// MultiHeadAttention_71176198029917
// MI455X (gfx1250) — compile-verified, matrix operand order checked by reading, not run on hardware
//
#include <hip/hip_runtime.h>

// ---------------- problem constants ----------------
#define Bc 4
#define Sc 2048
#define Dc 768
#define Hc 12
#define HDc 64
#define Mc (Bc * Sc)        // 8192 rows
#define N_QKV (3 * Dc)      // 2304

// ---------------- types ----------------
typedef __attribute__((ext_vector_type(16))) __bf16       v16bf;
typedef __attribute__((ext_vector_type(8)))  float        v8f;
typedef __attribute__((ext_vector_type(4)))  unsigned int u32x4;

union FragBF {
    v16bf v;
    u32x4 q[2];
};

__device__ __forceinline__ unsigned short f2bf(float f) {
    unsigned int u = __builtin_bit_cast(unsigned int, f);
    unsigned int r = u + 0x7FFFu + ((u >> 16) & 1u);   // round-to-nearest-even
    return (unsigned short)(r >> 16);
}

// Load a 16x32 bf16 A/B fragment for v_wmma_f32_16x16x32_bf16.
// Lanes 0-15 (g=0) hold K = [0..7] and [16..23]; lanes 16-31 (g=1) hold
// K = [8..15] and [24..31]. Each half is 8 contiguous halfwords = one
// 16-byte (b128) load.
__device__ __forceinline__ v16bf load_frag(const unsigned short* p, int kA) {
    FragBF f;
    f.q[0] = *(const u32x4*)(p + kA);
    f.q[1] = *(const u32x4*)(p + 16 + kA);
    return f.v;
}

__device__ __forceinline__ v8f wmma_bf16(v16bf a, v16bf b, v8f c) {
    return __builtin_amdgcn_wmma_f32_16x16x32_bf16(
        /*neg_a=*/false, a, /*neg_b=*/false, b,
        /*c_mod=*/(short)0, c, /*reuse_a=*/false, /*reuse_b=*/false);
}

// ---------------- conversion kernels ----------------
__global__ void cvt_f32_bf16(const float* __restrict__ src,
                             unsigned short* __restrict__ dst, int n) {
    int i = blockIdx.x * blockDim.x + threadIdx.x;
    if (i < n) dst[i] = f2bf(src[i]);
}

// src is (K rows x C cols) row-major fp32; dst is (C x K) row-major bf16.
__global__ void transpose_cvt(const float* __restrict__ src,
                              unsigned short* __restrict__ dst, int K, int C) {
    int i = blockIdx.x * blockDim.x + threadIdx.x;
    if (i >= K * C) return;
    int k = i / C, c = i % C;
    dst[(size_t)c * K + k] = f2bf(src[i]);
}

// ---------------- QKV projection GEMM ----------------
// X16: (8192 x 768) bf16, Wt: (2304 x 768) bf16 (transposed w_qkv).
// 32x128 output tile per wave (2 A-frags x 8 B-frags, 16 WMMAs per K-step).
// Scatter into Q/K (B,H,S,64) and Vt (B,H,64,S).
__global__ __launch_bounds__(32)
void qkv_gemm(const unsigned short* __restrict__ X,
              const unsigned short* __restrict__ Wt,
              const float* __restrict__ bias,
              unsigned short* __restrict__ Qb,
              unsigned short* __restrict__ Kb,
              unsigned short* __restrict__ Vt) {
    const int lane = threadIdx.x;
    const int lr = lane & 15, lg = lane >> 4;
    const int kA = lg * 8;
    const int mt = blockIdx.x * 32;
    const int nt = blockIdx.y * 128;

    v8f acc[2][8] = {};
    const unsigned short* a0 = X + (size_t)(mt + lr) * Dc;
    const unsigned short* a1 = X + (size_t)(mt + 16 + lr) * Dc;
    const unsigned short* bp[8];
#pragma unroll
    for (int c = 0; c < 8; ++c) bp[c] = Wt + (size_t)(nt + c * 16 + lr) * Dc;

    for (int k0 = 0; k0 < Dc; k0 += 32) {
        v16bf af0 = load_frag(a0 + k0, kA);
        v16bf af1 = load_frag(a1 + k0, kA);
#pragma unroll
        for (int c = 0; c < 8; ++c) {
            v16bf bf = load_frag(bp[c] + k0, kA);
            acc[0][c] = wmma_bf16(af0, bf, acc[0][c]);
            acc[1][c] = wmma_bf16(af1, bf, acc[1][c]);
        }
    }

    const int t = nt / Dc;   // 0=Q,1=K,2=V  (uniform per block: 768 % 128 == 0)
#pragma unroll
    for (int r = 0; r < 2; ++r) {
#pragma unroll
        for (int c = 0; c < 8; ++c) {
            const int colbase = nt + c * 16;
            const int h = (colbase % Dc) / HDc;       // head for this subtile
            const int hd = (colbase % HDc) + lr;      // 0..63
            const float bval = bias[colbase + lr];
#pragma unroll
            for (int j = 0; j < 8; ++j) {
                int gm = mt + r * 16 + j + 8 * lg;    // global row = b*S + s
                int b = gm >> 11, s = gm & (Sc - 1);
                unsigned short o = f2bf(acc[r][c][j] + bval);
                size_t bh = (size_t)(b * Hc + h);
                if (t == 0)      Qb[((bh * Sc + s) << 6) + hd] = o;
                else if (t == 1) Kb[((bh * Sc + s) << 6) + hd] = o;
                else             Vt[((bh << 6) + hd) * Sc + s] = o;
            }
        }
    }
}

// ---------------- flash attention (one wave per b,h,32-query tile) ----------
__global__ __launch_bounds__(32)
void attn_kernel(const unsigned short* __restrict__ Qb,
                 const unsigned short* __restrict__ Kb,
                 const unsigned short* __restrict__ Vt,
                 unsigned short* __restrict__ O) {
    __shared__ alignas(16) unsigned short pTile[2][16 * 32];  // P: 2x(16q x 32k)

    const int lane = threadIdx.x;
    const int lr = lane & 15, lg = lane >> 4;
    const int kA = lg * 8;
    const int qt = blockIdx.x;          // 32-query tile (0..63)
    const int h = blockIdx.y, b = blockIdx.z;
    const size_t bh = (size_t)(b * Hc + h);

    const unsigned short* qbase = Qb + ((bh * Sc + qt * 32) << 6);
    const unsigned short* kbase = Kb + (bh * Sc << 6);
    const unsigned short* vbase = Vt + ((bh << 6) * Sc);

    v16bf qf[2][2];
#pragma unroll
    for (int r = 0; r < 2; ++r) {
        qf[r][0] = load_frag(qbase + (r * 16 + lr) * HDc, kA);
        qf[r][1] = load_frag(qbase + (r * 16 + lr) * HDc + 32, kA);
    }

    float mrun[2][8], lrun[2][8];
#pragma unroll
    for (int r = 0; r < 2; ++r)
#pragma unroll
        for (int j = 0; j < 8; ++j) { mrun[r][j] = -1e30f; lrun[r][j] = 0.0f; }
    v8f acc[2][4] = {};

    const float scale = 0.125f;  // 1/sqrt(64)

    for (int kb0 = 0; kb0 < Sc; kb0 += 32) {
        // prefetch next key block rows and V columns into cache
        if (kb0 + 32 < Sc) {
            __builtin_prefetch(kbase + ((size_t)(kb0 + 32 + lane) << 6), 0, 3);
            __builtin_prefetch(vbase + (size_t)lane * Sc + kb0 + 32, 0, 3);
            __builtin_prefetch(vbase + (size_t)(lane + 32) * Sc + kb0 + 32, 0, 3);
        }
        // ---- scores: 32 queries x 32 keys, 8 WMMAs (K-frags reused x2) ----
        v8f sc[2][2] = {};
#pragma unroll
        for (int sub = 0; sub < 2; ++sub) {
            const unsigned short* krow =
                kbase + ((size_t)(kb0 + sub * 16 + lr) << 6);
            v16bf kf0 = load_frag(krow, kA);
            v16bf kf1 = load_frag(krow + 32, kA);
#pragma unroll
            for (int r = 0; r < 2; ++r) {
                sc[r][sub] = wmma_bf16(qf[r][0], kf0, sc[r][sub]);
                sc[r][sub] = wmma_bf16(qf[r][1], kf1, sc[r][sub]);
            }
        }
        // ---- online softmax (row = j + 8*lg, cols striped across 16 lanes) ----
#pragma unroll
        for (int r = 0; r < 2; ++r) {
#pragma unroll
            for (int j = 0; j < 8; ++j) {
                float s0 = sc[r][0][j] * scale;
                float s1 = sc[r][1][j] * scale;
                float mx = fmaxf(s0, s1);
#pragma unroll
                for (int off = 1; off < 16; off <<= 1)
                    mx = fmaxf(mx, __shfl_xor(mx, off, 32));
                float mnew = fmaxf(mrun[r][j], mx);
                float corr = __expf(mrun[r][j] - mnew);
                float p0 = __expf(s0 - mnew);
                float p1 = __expf(s1 - mnew);
                float ps = p0 + p1;
#pragma unroll
                for (int off = 1; off < 16; off <<= 1)
                    ps += __shfl_xor(ps, off, 32);
                lrun[r][j] = lrun[r][j] * corr + ps;
                mrun[r][j] = mnew;
                acc[r][0][j] *= corr; acc[r][1][j] *= corr;
                acc[r][2][j] *= corr; acc[r][3][j] *= corr;
                pTile[r][(j + 8 * lg) * 32 + lr] = f2bf(p0);
                pTile[r][(j + 8 * lg) * 32 + 16 + lr] = f2bf(p1);
            }
        }
        __syncthreads();
        // ---- PV: P(32x32) @ V(32x64), 8 WMMAs (V-frags reused x2) ----
        v16bf pf0 = load_frag(&pTile[0][lr * 32], kA);
        v16bf pf1 = load_frag(&pTile[1][lr * 32], kA);
#pragma unroll
        for (int c = 0; c < 4; ++c) {
            v16bf vf = load_frag(vbase + (size_t)(c * 16 + lr) * Sc + kb0, kA);
            acc[0][c] = wmma_bf16(pf0, vf, acc[0][c]);
            acc[1][c] = wmma_bf16(pf1, vf, acc[1][c]);
        }
        __syncthreads();
    }

    // ---- normalize + store concat-head layout (B*S, 768) in bf16 ----
    const size_t orow0 = (size_t)b * Sc + qt * 32;
#pragma unroll
    for (int r = 0; r < 2; ++r) {
#pragma unroll
        for (int c = 0; c < 4; ++c) {
#pragma unroll
            for (int j = 0; j < 8; ++j) {
                int m = r * 16 + j + 8 * lg;
                O[(orow0 + m) * Dc + h * HDc + c * 16 + lr] =
                    f2bf(acc[r][c][j] / lrun[r][j]);
            }
        }
    }
}

// ---------------- output projection GEMM ----------------
// A: (8192 x 768) bf16, Wt: (768 x 768) bf16 (transposed w_proj). fp32 out.
__global__ __launch_bounds__(32)
void proj_gemm(const unsigned short* __restrict__ A,
               const unsigned short* __restrict__ Wt,
               const float* __restrict__ bias,
               float* __restrict__ out) {
    const int lane = threadIdx.x;
    const int lr = lane & 15, lg = lane >> 4;
    const int kA = lg * 8;
    const int mt = blockIdx.x * 32;
    const int nt = blockIdx.y * 128;

    v8f acc[2][8] = {};
    const unsigned short* a0 = A + (size_t)(mt + lr) * Dc;
    const unsigned short* a1 = A + (size_t)(mt + 16 + lr) * Dc;
    const unsigned short* bp[8];
#pragma unroll
    for (int c = 0; c < 8; ++c) bp[c] = Wt + (size_t)(nt + c * 16 + lr) * Dc;

    for (int k0 = 0; k0 < Dc; k0 += 32) {
        v16bf af0 = load_frag(a0 + k0, kA);
        v16bf af1 = load_frag(a1 + k0, kA);
#pragma unroll
        for (int c = 0; c < 8; ++c) {
            v16bf bf = load_frag(bp[c] + k0, kA);
            acc[0][c] = wmma_bf16(af0, bf, acc[0][c]);
            acc[1][c] = wmma_bf16(af1, bf, acc[1][c]);
        }
    }
#pragma unroll
    for (int r = 0; r < 2; ++r) {
#pragma unroll
        for (int c = 0; c < 8; ++c) {
            const int col = nt + c * 16 + lr;
            const float bval = bias[col];
#pragma unroll
            for (int j = 0; j < 8; ++j) {
                int gm = mt + r * 16 + j + 8 * lg;
                out[(size_t)gm * Dc + col] = acc[r][c][j] + bval;
            }
        }
    }
}

// ---------------- launcher ----------------
extern "C" void kernel_launch(void* const* d_in, const int* in_sizes, int n_in,
                              void* d_out, int out_size, void* d_ws, size_t ws_size,
                              hipStream_t stream) {
    const float* x      = (const float*)d_in[0];
    const float* w_qkv  = (const float*)d_in[1];
    const float* b_qkv  = (const float*)d_in[2];
    const float* w_proj = (const float*)d_in[3];
    const float* b_proj = (const float*)d_in[4];
    float* out = (float*)d_out;

    // workspace layout (bf16 stored as unsigned short)
    char* ws = (char*)d_ws;
    unsigned short* X16    = (unsigned short*)ws; ws += (size_t)Mc * Dc * 2;
    unsigned short* WqkvT  = (unsigned short*)ws; ws += (size_t)N_QKV * Dc * 2;
    unsigned short* WprojT = (unsigned short*)ws; ws += (size_t)Dc * Dc * 2;
    unsigned short* Qb     = (unsigned short*)ws; ws += (size_t)Mc * Dc * 2;
    unsigned short* Kb     = (unsigned short*)ws; ws += (size_t)Mc * Dc * 2;
    unsigned short* Vt     = (unsigned short*)ws; ws += (size_t)Mc * Dc * 2;
    unsigned short* Ob     = (unsigned short*)ws; ws += (size_t)Mc * Dc * 2;

    // 1) precision conversions / weight transposes
    {
        int n = Mc * Dc;
        cvt_f32_bf16<<<(n + 255) / 256, 256, 0, stream>>>(x, X16, n);
        int nw = Dc * N_QKV;
        transpose_cvt<<<(nw + 255) / 256, 256, 0, stream>>>(w_qkv, WqkvT, Dc, N_QKV);
        int np = Dc * Dc;
        transpose_cvt<<<(np + 255) / 256, 256, 0, stream>>>(w_proj, WprojT, Dc, Dc);
    }
    // 2) fused QKV projection (+bias), scatter to Q/K/(V transposed)
    qkv_gemm<<<dim3(Mc / 32, N_QKV / 128), 32, 0, stream>>>(X16, WqkvT, b_qkv,
                                                            Qb, Kb, Vt);
    // 3) flash attention per (b, h, 32-query tile)
    attn_kernel<<<dim3(Sc / 32, Hc, Bc), 32, 0, stream>>>(Qb, Kb, Vt, Ob);
    // 4) output projection (+bias) to fp32
    proj_gemm<<<dim3(Mc / 32, Dc / 128), 32, 0, stream>>>(Ob, WprojT, b_proj, out);
}
